// FP8Linear_79001628442991
// MI455X (gfx1250) — 8-bit matrix operand order repaired and checked by reading, compiled, not run on hardware
//
#include <hip/hip_runtime.h>
#include <stdint.h>


typedef int   v16i __attribute__((ext_vector_type(16)));
typedef float v8f  __attribute__((ext_vector_type(8)));

#define M_DIM   8192
#define N_DIM   2048
#define K_DIM   2048
#define KBLK    (K_DIM / 128)
#define FP8MAX  448.0f

#define ROW_STRIDE 144
#define A_BYTES    (128 * ROW_STRIDE)
#define B_BYTES    (64 * ROW_STRIDE)
#define STAGE_BYTES (A_BYTES + B_BYTES)

__device__ __forceinline__ unsigned f32_to_e4m3_sw(float f) {
  unsigned u    = __float_as_uint(f);
  unsigned sign = (u >> 24) & 0x80u;
  float a = fminf(fabsf(f), FP8MAX);
  unsigned b   = __float_as_uint(a);
  unsigned exp = b >> 23;
  unsigned byte;
  if (exp >= 121u) {
    unsigned mant = (b >> 20) & 7u;
    unsigned rem  = b & 0xFFFFFu;
    unsigned inc  = (rem > 0x80000u) || (rem == 0x80000u && (mant & 1u));
    unsigned em   = (((exp - 120u) << 3) | mant) + inc;
    if (em > 0x7Eu) em = 0x7Eu;
    byte = em;
  } else if (exp >= 110u) {
    byte = (unsigned)(int)rintf(a * 512.0f);
  } else {
    byte = 0u;
  }
  return sign | byte;
}

__device__ __forceinline__ unsigned pack4_e4m3(float a0, float a1, float a2, float a3) {
#if __has_builtin(__builtin_amdgcn_cvt_pk_fp8_f32)
  int lo   = __builtin_amdgcn_cvt_pk_fp8_f32(a0, a1, 0,  false);
  int full = __builtin_amdgcn_cvt_pk_fp8_f32(a2, a3, lo, true);
  return (unsigned)full;
#else
  return f32_to_e4m3_sw(a0) | (f32_to_e4m3_sw(a1) << 8) |
         (f32_to_e4m3_sw(a2) << 16) | (f32_to_e4m3_sw(a3) << 24);
#endif
}

__device__ __forceinline__ void async_copy16(unsigned lds_off, const uint8_t* gbase,
                                             unsigned goff) {
  asm volatile("global_load_async_to_lds_b128 %0, %1, %2"
               :: "v"(lds_off), "v"(goff), "s"(gbase) : "memory");
}
__device__ __forceinline__ void wait_async0() {
  asm volatile("s_wait_asynccnt 0x0" ::: "memory");
}

__global__ void __launch_bounds__(256) quant_x_kernel(const float* __restrict__ x,
                                                      uint8_t* __restrict__ x8,
                                                      float* __restrict__ xs) {
  int wave = threadIdx.x >> 5;
  int lane = threadIdx.x & 31;
  int chunk = blockIdx.x * 8 + wave;
  int row = chunk >> 4;
  int cb  = chunk & 15;
  const float* p = x + (size_t)row * K_DIM + cb * 128 + lane * 4;
  float4 v = *(const float4*)p;
  float am = fmaxf(fmaxf(fabsf(v.x), fabsf(v.y)), fmaxf(fabsf(v.z), fabsf(v.w)));
  #pragma unroll
  for (int off = 16; off > 0; off >>= 1)
    am = fmaxf(am, __shfl_xor(am, off, 32));
  am = fmaxf(am, 1e-4f);
  float s = FP8MAX / am;
  unsigned q = pack4_e4m3(v.x * s, v.y * s, v.z * s, v.w * s);
  *(unsigned*)(x8 + (size_t)row * K_DIM + cb * 128 + lane * 4) = q;
  if (lane == 0) xs[cb * M_DIM + row] = am / FP8MAX;
}

__global__ void __launch_bounds__(256) quant_w_kernel(const float* __restrict__ w,
                                                      uint8_t* __restrict__ w8,
                                                      float* __restrict__ wsc) {
  __shared__ float red[256];
  int t  = threadIdx.x;
  int br = blockIdx.x >> 4;
  int bc = blockIdx.x & 15;
  int row  = t >> 1;
  int col0 = (t & 1) * 64;
  const float* p = w + (size_t)(br * 128 + row) * K_DIM + bc * 128 + col0;
  float am = 0.0f;
  #pragma unroll
  for (int i = 0; i < 16; ++i) {
    float4 v = ((const float4*)p)[i];
    am = fmaxf(am, fmaxf(fmaxf(fabsf(v.x), fabsf(v.y)), fmaxf(fabsf(v.z), fabsf(v.w))));
  }
  red[t] = am;
  __syncthreads();
  #pragma unroll
  for (int s = 128; s > 0; s >>= 1) {
    if (t < s) red[t] = fmaxf(red[t], red[t + s]);
    __syncthreads();
  }
  float amax = fmaxf(red[0], 1e-4f);
  float s = FP8MAX / amax;
  uint8_t* q8 = w8 + (size_t)(br * 128 + row) * K_DIM + bc * 128 + col0;
  #pragma unroll
  for (int i = 0; i < 16; ++i) {
    float4 v = ((const float4*)p)[i];
    *(unsigned*)(q8 + i * 4) = pack4_e4m3(v.x * s, v.y * s, v.z * s, v.w * s);
  }
  if (t == 0) wsc[blockIdx.x] = amax / FP8MAX;
}

__global__ void __launch_bounds__(256) gemm_fp8_kernel(const uint8_t* __restrict__ x8,
                                                       const uint8_t* __restrict__ w8,
                                                       const float* __restrict__ xs,
                                                       const float* __restrict__ wsc,
                                                       unsigned short* __restrict__ out) {
  __shared__ __attribute__((aligned(16))) uint8_t smem[2 * STAGE_BYTES];

  const int t    = threadIdx.x;
  const int wave = t >> 5, lane = t & 31;
  const int wm = wave & 3, wn = wave >> 2;
  const int l16 = lane & 15, hi = lane >> 4;
  const int Mwg = blockIdx.y * 128;
  const int Nwg = blockIdx.x * 64;
  const int nb  = blockIdx.x >> 1;

  const unsigned lds0 = (unsigned)(uintptr_t)&smem[0];

  const int arow = t >> 1;
  const int ahalf = (t & 1) * 64;
  const int bcol = t >> 2;
  const int bseg = (t & 3) * 32;

  const unsigned agoff_base = (unsigned)((Mwg + arow) * K_DIM + ahalf);
  const unsigned bgoff_base = (unsigned)((Nwg + bcol) * K_DIM + bseg);
  const unsigned alds_base  = lds0 + arow * ROW_STRIDE + ahalf;
  const unsigned blds_base  = lds0 + A_BYTES + bcol * ROW_STRIDE + bseg;

  auto issue_stage = [&](int stage, int kb) {
    const unsigned soff = (unsigned)(stage * STAGE_BYTES);
    const unsigned k0   = (unsigned)(kb * 128);
    #pragma unroll
    for (int j = 0; j < 4; ++j)
      async_copy16(alds_base + soff + j * 16, x8, agoff_base + k0 + j * 16);
    #pragma unroll
    for (int j = 0; j < 2; ++j)
      async_copy16(blds_base + soff + j * 16, w8, bgoff_base + k0 + j * 16);
  };

  v8f acc[2][2];
  #pragma unroll
  for (int mt = 0; mt < 2; ++mt)
    #pragma unroll
    for (int nt = 0; nt < 2; ++nt)
      acc[mt][nt] = (v8f){0.f, 0.f, 0.f, 0.f, 0.f, 0.f, 0.f, 0.f};

  issue_stage(0, 0);

  for (int kb = 0; kb < KBLK; ++kb) {
    const int cur = kb & 1;
    wait_async0();
    __syncthreads();
    if (kb + 1 < KBLK) issue_stage(1 - cur, kb + 1);

    const uint8_t* sa = smem + cur * STAGE_BYTES;
    const uint8_t* sb = smem + cur * STAGE_BYTES + A_BYTES;

    v16i A[2];
    #pragma unroll
    for (int mt = 0; mt < 2; ++mt) {
      const uint8_t* pa = sa + (wm * 32 + mt * 16 + l16) * ROW_STRIDE + hi * 8;
      v16i a;
      #pragma unroll
      for (int j = 0; j < 4; ++j) {
        uint2 t0 = *(const uint2*)(pa + 16 * j);
        a[2 * j]     = (int)t0.x;
        a[2 * j + 1] = (int)t0.y;
        uint2 t1 = *(const uint2*)(pa + 64 + 16 * j);
        a[8 + 2 * j] = (int)t1.x;
        a[9 + 2 * j] = (int)t1.y;
      }
      A[mt] = a;
    }

    float wsv = wsc[nb * KBLK + kb];
    float xsv[2][8];
    #pragma unroll
    for (int mt = 0; mt < 2; ++mt) {
      const float* pxs = xs + (size_t)kb * M_DIM + Mwg + wm * 32 + mt * 16 + 8 * hi;
      float4 s0 = ((const float4*)pxs)[0];
      float4 s1 = ((const float4*)pxs)[1];
      xsv[mt][0] = s0.x; xsv[mt][1] = s0.y; xsv[mt][2] = s0.z; xsv[mt][3] = s0.w;
      xsv[mt][4] = s1.x; xsv[mt][5] = s1.y; xsv[mt][6] = s1.z; xsv[mt][7] = s1.w;
    }

    #pragma unroll
    for (int nt = 0; nt < 2; ++nt) {
      const uint8_t* pb = sb + (wn * 32 + nt * 16 + l16) * ROW_STRIDE + hi * 8;
      v16i b;
      #pragma unroll
      for (int j = 0; j < 4; ++j) {
        uint2 tb0 = *(const uint2*)(pb + 32 * j);
        uint2 tb1 = *(const uint2*)(pb + 32 * j + 16);
        b[4 * j]     = (int)tb0.x;
        b[4 * j + 1] = (int)tb0.y;
        b[4 * j + 2] = (int)tb1.x;
        b[4 * j + 3] = (int)tb1.y;
      }
      #pragma unroll
      for (int mt = 0; mt < 2; ++mt) {
        v8f z = (v8f){0.f, 0.f, 0.f, 0.f, 0.f, 0.f, 0.f, 0.f};
        v8f c = __builtin_amdgcn_wmma_f32_16x16x128_fp8_fp8(
            A[mt], b, (short)0, z, false, false);
        #pragma unroll
        for (int i = 0; i < 8; ++i)
          acc[mt][nt][i] += c[i] * (xsv[mt][i] * wsv);
      }
    }
  }

  #pragma unroll
  for (int mt = 0; mt < 2; ++mt)
    #pragma unroll
    for (int nt = 0; nt < 2; ++nt)
      #pragma unroll
      for (int i = 0; i < 8; ++i) {
        int m = Mwg + wm * 32 + mt * 16 + 8 * hi + i;
        int n = Nwg + wn * 32 + nt * 16 + l16;
        unsigned u = __float_as_uint(acc[mt][nt][i]);
        u += 0x7FFFu + ((u >> 16) & 1u);
        out[(size_t)m * N_DIM + n] = (unsigned short)(u >> 16);
      }
}

extern "C" void kernel_launch(void* const* d_in, const int* in_sizes, int n_in,
                              void* d_out, int out_size, void* d_ws, size_t ws_size,
                              hipStream_t stream) {
  const float* x = (const float*)d_in[0];
  const float* w = (const float*)d_in[1];

  uint8_t* base = (uint8_t*)d_ws;
  uint8_t* x8  = base;
  uint8_t* w8  = x8 + (size_t)M_DIM * K_DIM;
  float*   xsc = (float*)(w8 + (size_t)N_DIM * K_DIM);
  float*   wsc = xsc + (size_t)KBLK * M_DIM;
  unsigned short* out = (unsigned short*)d_out;

  quant_x_kernel<<<dim3(M_DIM * KBLK / 8), 256, 0, stream>>>(x, x8, xsc);
  quant_w_kernel<<<dim3((N_DIM / 128) * (K_DIM / 128)), 256, 0, stream>>>(w, w8, wsc);
  gemm_fp8_kernel<<<dim3(N_DIM / 64, M_DIM / 128), 256, 0, stream>>>(x8, w8, xsc, wsc, out);
}
